// BiLSTM_GCN_2_45715631899084
// MI455X (gfx1250) — compile-verified, matrix operand order checked by reading, not run on hardware
//
#include <hip/hip_runtime.h>

// ---------------- types / WMMA helpers ----------------
typedef __attribute__((ext_vector_type(16))) __bf16 v16bf;
typedef __attribute__((ext_vector_type(8)))  float  v8f;

union FragU { v16bf v; uint4 q[2]; };

__device__ __forceinline__ v8f zero_v8f() {
  v8f z;
#pragma unroll
  for (int i = 0; i < 8; ++i) z[i] = 0.0f;
  return z;
}

__device__ __forceinline__ v8f wmma_bf16(v16bf a, v16bf b, v8f c) {
  // D = A(16x32 bf16) * B(32x16 bf16) + C(16x16 f32)
  return __builtin_amdgcn_wmma_f32_16x16x32_bf16(
      /*neg_a=*/false, a, /*neg_b=*/false, b,
      /*c_mod=*/(short)0, c, /*reuse_a=*/false, /*reuse_b=*/false);
}

// Per-lane fragment load, row-major bf16 source with leading dim ld.
// A fragment: lane&15 = row (M), lane>>4 selects K-halves:
//   elems 0..7  -> k = k0 + half*8 + j
//   elems 8..15 -> k = k0 + 16 + half*8 + j
// B fragment (weights stored [N,K] row-major) uses identical addressing with
// lane&15 = output column N.
__device__ __forceinline__ v16bf load_frag(const __bf16* __restrict__ p,
                                           int ld, int row0, int k0) {
  const int lane = threadIdx.x & 31;
  const int m = lane & 15, half = lane >> 4;
  const __bf16* q = p + (long)(row0 + m) * ld + k0 + half * 8;
  FragU u;
  u.q[0] = *(const uint4*)(q);
  u.q[1] = *(const uint4*)(q + 16);   // +16 bf16 elements
  return u.v;
}

__device__ __forceinline__ void prefetch_frag(const __bf16* __restrict__ p,
                                              int ld, int row0, int k0) {
  const int lane = threadIdx.x & 31;
  const int m = lane & 15, half = lane >> 4;
  __builtin_prefetch(p + (long)(row0 + m) * ld + k0 + half * 8, 0, 0);
}

// ---------------- GEMM: C[M,N] = A[M,K](bf16) * B[N,K]^T(bf16) + bias ----------------
// grid: (M/64, N/128), block: 256 threads (8 waves). Wave tile = 16x64.
// Double-buffered fragments: while WMMAs consume step k, loads for k+32 are in
// flight (avoids s_wait_loadcnt 0 before every WMMA).
__global__ void __launch_bounds__(256)
gemm_bf16(const __bf16* __restrict__ A, const __bf16* __restrict__ B,
          float* __restrict__ C, const float* __restrict__ bias,
          int M, int N, int K, int do_relu) {
  const int wid  = threadIdx.x >> 5;
  const int row0 = blockIdx.x * 64  + (wid & 3) * 16;
  const int col0 = blockIdx.y * 128 + (wid >> 2) * 64;

  v8f acc[4];
#pragma unroll
  for (int u = 0; u < 4; ++u) acc[u] = zero_v8f();

  // prologue: fragments for k = 0
  v16bf a_cur = load_frag(A, K, row0, 0);
  v16bf b_cur0 = load_frag(B, K, col0 +  0, 0);
  v16bf b_cur1 = load_frag(B, K, col0 + 16, 0);
  v16bf b_cur2 = load_frag(B, K, col0 + 32, 0);
  v16bf b_cur3 = load_frag(B, K, col0 + 48, 0);

  for (int k0 = 32; k0 < K; k0 += 32) {
    // issue next-step loads first so they overlap the WMMAs below
    v16bf a_nxt = load_frag(A, K, row0, k0);
    v16bf b_nxt0 = load_frag(B, K, col0 +  0, k0);
    v16bf b_nxt1 = load_frag(B, K, col0 + 16, k0);
    v16bf b_nxt2 = load_frag(B, K, col0 + 32, k0);
    v16bf b_nxt3 = load_frag(B, K, col0 + 48, k0);
    prefetch_frag(A, K, row0, k0 + 32);   // stream A ahead (global_prefetch_b8)

    acc[0] = wmma_bf16(a_cur, b_cur0, acc[0]);
    acc[1] = wmma_bf16(a_cur, b_cur1, acc[1]);
    acc[2] = wmma_bf16(a_cur, b_cur2, acc[2]);
    acc[3] = wmma_bf16(a_cur, b_cur3, acc[3]);

    a_cur = a_nxt;
    b_cur0 = b_nxt0; b_cur1 = b_nxt1; b_cur2 = b_nxt2; b_cur3 = b_nxt3;
  }
  acc[0] = wmma_bf16(a_cur, b_cur0, acc[0]);
  acc[1] = wmma_bf16(a_cur, b_cur1, acc[1]);
  acc[2] = wmma_bf16(a_cur, b_cur2, acc[2]);
  acc[3] = wmma_bf16(a_cur, b_cur3, acc[3]);

  const int lane = threadIdx.x & 31;
  const int n_ = lane & 15, half = lane >> 4;
#pragma unroll
  for (int u = 0; u < 4; ++u) {
    const int n = col0 + u * 16 + n_;
    const float bv = bias ? bias[n] : 0.0f;
#pragma unroll
    for (int r = 0; r < 8; ++r) {
      float v = acc[u][r] + bv;
      if (do_relu) v = fmaxf(v, 0.0f);
      C[(long)(row0 + r + 8 * half) * N + n] = v;
    }
  }
}

// ---------------- LSTM recurrent scan ----------------
// One block = 16 batch rows, 16 waves. Wave w owns hidden cols [16w,16w+16)
// for ALL FOUR gates (cols 16w, 16w+256, 16w+512, 16w+768) so gate math stays
// in registers; c-state lives in a v8f register across all 35 steps.
// xg: [2048*35, 1024] f32 (node = b*35+t), whh: [1024,256] bf16,
// hout: [2048*35, 512] f32 written at column offset coff.
__global__ void __launch_bounds__(512)
lstm_scan(const float* __restrict__ xg, const __bf16* __restrict__ whh,
          float* __restrict__ hout, int reverse, int coff) {
  __shared__ __align__(16) __bf16 hbf[16 * 256];

  const int tid  = threadIdx.x;
  const int wid  = tid >> 5;
  const int lane = tid & 31;
  const int n_   = lane & 15;
  const int half = lane >> 4;
  const int b0   = blockIdx.x * 16;
  const int colbase = wid * 16;
  const int hh = colbase + n_;

  for (int i = tid; i < 16 * 256; i += 512) hbf[i] = (__bf16)0.0f;
  v8f c_state = zero_v8f();
  __syncthreads();

  for (int s = 0; s < 35; ++s) {
    const int t = reverse ? (34 - s) : s;

    v8f acc[4];
#pragma unroll
    for (int g = 0; g < 4; ++g) acc[g] = zero_v8f();

    // double-buffered K-loop over the recurrent GEMM h @ W_hh^T
    v16bf a_cur = load_frag(hbf, 256, 0, 0);
    v16bf b_cur0 = load_frag(whh, 256, colbase +   0, 0);
    v16bf b_cur1 = load_frag(whh, 256, colbase + 256, 0);
    v16bf b_cur2 = load_frag(whh, 256, colbase + 512, 0);
    v16bf b_cur3 = load_frag(whh, 256, colbase + 768, 0);
#pragma unroll
    for (int k0 = 32; k0 < 256; k0 += 32) {
      v16bf a_nxt = load_frag(hbf, 256, 0, k0);
      v16bf b_nxt0 = load_frag(whh, 256, colbase +   0, k0);
      v16bf b_nxt1 = load_frag(whh, 256, colbase + 256, k0);
      v16bf b_nxt2 = load_frag(whh, 256, colbase + 512, k0);
      v16bf b_nxt3 = load_frag(whh, 256, colbase + 768, k0);

      acc[0] = wmma_bf16(a_cur, b_cur0, acc[0]);
      acc[1] = wmma_bf16(a_cur, b_cur1, acc[1]);
      acc[2] = wmma_bf16(a_cur, b_cur2, acc[2]);
      acc[3] = wmma_bf16(a_cur, b_cur3, acc[3]);

      a_cur = a_nxt;
      b_cur0 = b_nxt0; b_cur1 = b_nxt1; b_cur2 = b_nxt2; b_cur3 = b_nxt3;
    }
    acc[0] = wmma_bf16(a_cur, b_cur0, acc[0]);
    acc[1] = wmma_bf16(a_cur, b_cur1, acc[1]);
    acc[2] = wmma_bf16(a_cur, b_cur2, acc[2]);
    acc[3] = wmma_bf16(a_cur, b_cur3, acc[3]);

    __syncthreads();   // all waves done reading hbf

#pragma unroll
    for (int r = 0; r < 8; ++r) {
      const int m = r + 8 * half;
      const long row = (long)(b0 + m) * 35 + t;
      const float* xr = xg + row * 1024;
      const float gi = acc[0][r] + xr[hh];
      const float gf = acc[1][r] + xr[256 + hh];
      const float gg = acc[2][r] + xr[512 + hh];
      const float go = acc[3][r] + xr[768 + hh];
      const float si = 1.0f / (1.0f + expf(-gi));
      const float sf = 1.0f / (1.0f + expf(-gf));
      const float so = 1.0f / (1.0f + expf(-go));
      const float c  = sf * c_state[r] + si * tanhf(gg);
      const float h  = so * tanhf(c);
      c_state[r] = c;
      hbf[m * 256 + hh] = (__bf16)h;
      hout[row * 512 + coff + hh] = h;
    }
    __syncthreads();   // h published before next step's fragment reads
  }
}

// ---------------- elementwise / graph kernels ----------------
__global__ void zero_f32(float* __restrict__ p, long n) {
  long i = (long)blockIdx.x * blockDim.x + threadIdx.x;
  if (i < n) p[i] = 0.0f;
}

__global__ void add_vec(const float* __restrict__ a, const float* __restrict__ b,
                        float* __restrict__ c, int n) {
  int i = blockIdx.x * blockDim.x + threadIdx.x;
  if (i < n) c[i] = a[i] + b[i];
}

__global__ void cvt_pad_bf16(const float* __restrict__ x, __bf16* __restrict__ y,
                             long rows, int K, int Kp) {
  long i = (long)blockIdx.x * blockDim.x + threadIdx.x;
  long total = rows * Kp;
  if (i >= total) return;
  long r = i / Kp;
  int k = (int)(i - r * Kp);
  y[i] = (k < K) ? (__bf16)x[r * K + k] : (__bf16)0.0f;
}

__global__ void cvt_transpose_bf16(const float* __restrict__ w, __bf16* __restrict__ y,
                                   int K, int N) {  // w:[K,N] -> y:[N,K]
  long i = (long)blockIdx.x * blockDim.x + threadIdx.x;
  long total = (long)K * N;
  if (i >= total) return;
  int n = (int)(i / K);
  int k = (int)(i - (long)n * K);
  y[i] = (__bf16)w[(long)k * N + n];
}

__global__ void scale_cvt_bf16(const float* __restrict__ x, const float* __restrict__ rs,
                               __bf16* __restrict__ y, long rows, int cols) {
  long i = (long)blockIdx.x * blockDim.x + threadIdx.x;
  if (i >= rows * (long)cols) return;
  long r = i / cols;
  y[i] = (__bf16)(x[i] * rs[r]);
}

__global__ void deg_accum(const int* __restrict__ src, const int* __restrict__ dst,
                          float* __restrict__ dout, float* __restrict__ din, int E) {
  int e = blockIdx.x * blockDim.x + threadIdx.x;
  if (e < E) {
    atomicAdd(&dout[src[e]], 1.0f);
    atomicAdd(&din[dst[e]], 1.0f);
  }
}

__global__ void deg_rsqrt(float* __restrict__ d, long n) {
  long i = (long)blockIdx.x * blockDim.x + threadIdx.x;
  if (i < n) d[i] = rsqrtf(fmaxf(d[i], 1.0f));
}

__global__ void scatter_add(const int* __restrict__ src, const int* __restrict__ dst,
                            const float* __restrict__ h, float* __restrict__ agg, int E) {
  long i = (long)blockIdx.x * blockDim.x + threadIdx.x;
  if (i >= (long)E * 64) return;
  int e = (int)(i >> 6);
  int j = (int)(i & 63) * 4;
  const float4 v = *(const float4*)(h + (long)src[e] * 256 + j);
  float* a = agg + (long)dst[e] * 256 + j;
  atomicAdd(a + 0, v.x);
  atomicAdd(a + 1, v.y);
  atomicAdd(a + 2, v.z);
  atomicAdd(a + 3, v.w);
}

__global__ void gcn_combine(const float* __restrict__ agg, const float* __restrict__ rin,
                            const float* __restrict__ bias, const float* __restrict__ l2g,
                            float* __restrict__ out, long n) {
  long i = (long)blockIdx.x * blockDim.x + threadIdx.x;
  if (i >= n) return;
  long row = i >> 8;
  int c = (int)(i & 255);
  float v = agg[i] * rin[row] + bias[c];
  v = fmaxf(v, 0.0f);
  out[i] = 0.6f * v + 0.4f * l2g[i];
}

__global__ void graph_mean(const float* __restrict__ h, float* __restrict__ rep) {
  int i = blockIdx.x * blockDim.x + threadIdx.x;
  if (i >= 2048 * 256) return;
  int g = i >> 8, c = i & 255;
  float s = 0.0f;
  for (int t = 0; t < 35; ++t) s += h[((long)g * 35 + t) * 256 + c];
  rep[i] = s * (1.0f / 35.0f);
}

__global__ void pred_kernel(const float* __restrict__ r1, const float* __restrict__ r2,
                            const float* __restrict__ pw, const float* __restrict__ pb,
                            float* __restrict__ out) {
  int g = blockIdx.x * blockDim.x + threadIdx.x;
  if (g >= 2048) return;
  float a0 = pb[0], a1 = pb[1];
  for (int j = 0; j < 256; ++j) {
    float d = tanhf(r1[g * 256 + j] - r2[g * 256 + j]);
    a0 += d * pw[j];
    a1 += d * pw[256 + j];
  }
  out[g * 2 + 0] = a0;
  out[g * 2 + 1] = a1;
}

// ---------------- host orchestration ----------------
static inline int cdiv(long a, int b) { return (int)((a + b - 1) / b); }

extern "C" void kernel_launch(void* const* d_in, const int* in_sizes, int n_in,
                              void* d_out, int out_size, void* d_ws, size_t ws_size,
                              hipStream_t stream) {
  (void)n_in; (void)out_size; (void)ws_size;
  const int NN = 71680;       // nodes
  const int BGN = 2048;       // graphs

  const float* xin[2]  = { (const float*)d_in[0], (const float*)d_in[1] };
  const int*   srcs[2] = { (const int*)d_in[2], (const int*)d_in[4] };
  const int*   dsts[2] = { (const int*)d_in[3], (const int*)d_in[5] };
  const int E = in_sizes[2];

  const float* lin_w = (const float*)d_in[38];
  const float* lin_b = (const float*)d_in[39];
  const float* gw1[2] = { (const float*)d_in[40], (const float*)d_in[42] };
  const float* gb1[2] = { (const float*)d_in[41], (const float*)d_in[43] };
  const float* gw2[2] = { (const float*)d_in[44], (const float*)d_in[46] };
  const float* gb2[2] = { (const float*)d_in[45], (const float*)d_in[47] };
  const float* pred_w = (const float*)d_in[48];
  const float* pred_b = (const float*)d_in[49];

  // LSTM param resolver: handles insertion-order vs sorted-key flattening.
  const bool sorted_keys = (in_sizes[6] == 1024);
  auto lptr = [&](int br, int l, int rev, int kind) -> const float* {
    // kind: 0=w_ih, 1=w_hh, 2=b_ih, 3=b_hh
    const int base = 6 + 16 * br;
    int idx;
    if (!sorted_keys) {
      idx = base + l * 8 + rev * 4 + kind;
    } else {
      const int kmap[4] = {12, 8, 4, 0};  // w_ih, w_hh, b_ih, b_hh group starts
      idx = base + kmap[kind] + l * 2 + rev;
    }
    return (const float*)d_in[idx];
  };

  // workspace arena
  size_t off = 0;
  char* wsb = (char*)d_ws;
  auto alloc = [&](size_t bytes) -> void* {
    off = (off + 255) & ~(size_t)255;
    void* p = wsb + off;
    off += bytes;
    return p;
  };
  float*  XG   = (float*) alloc((size_t)NN * 1024 * 4);   // gate preactivations
  __bf16* ABUF = (__bf16*)alloc((size_t)NN * 512 * 2);    // bf16 activations (GEMM A)
  float*  LOUT = (float*) alloc((size_t)NN * 512 * 4);    // layer output / wf
  float*  L2G  = (float*) alloc((size_t)NN * 256 * 4);
  float*  T6   = (float*) alloc((size_t)NN * 256 * 4);
  float*  T7   = (float*) alloc((size_t)NN * 256 * 4);
  __bf16* WIH  = (__bf16*)alloc((size_t)1024 * 512 * 2);
  __bf16* WHH  = (__bf16*)alloc((size_t)1024 * 256 * 2);
  __bf16* WT   = (__bf16*)alloc((size_t)256 * 512 * 2);
  float*  BIASF= (float*) alloc(1024 * 4);
  float*  DEG  = (float*) alloc((size_t)2 * NN * 4);
  float*  REP[2];
  REP[0] = (float*)alloc((size_t)BGN * 256 * 4);
  REP[1] = (float*)alloc((size_t)BGN * 256 * 4);

  for (int br = 0; br < 2; ++br) {
    const int* src = srcs[br];
    const int* dst = dsts[br];
    float* DEGO = DEG;
    float* DEGI = DEG + NN;

    // degrees -> rsqrt(clip(deg,1))
    zero_f32<<<cdiv(2L * NN, 256), 256, 0, stream>>>(DEG, 2L * NN);
    deg_accum<<<cdiv(E, 256), 256, 0, stream>>>(src, dst, DEGO, DEGI, E);
    deg_rsqrt<<<cdiv(2L * NN, 256), 256, 0, stream>>>(DEG, 2L * NN);

    // layer 0 input: pad 300 -> 320 and convert to bf16
    cvt_pad_bf16<<<cdiv((long)NN * 320, 256), 256, 0, stream>>>(xin[br], ABUF, NN, 300, 320);

    for (int l = 0; l < 2; ++l) {
      const int Kin = l ? 512 : 300;
      const int Kp  = l ? 512 : 320;
      if (l == 1)
        cvt_pad_bf16<<<cdiv((long)NN * 512, 256), 256, 0, stream>>>(LOUT, ABUF, NN, 512, 512);
      for (int rev = 0; rev < 2; ++rev) {
        cvt_pad_bf16<<<cdiv((long)1024 * Kp, 256), 256, 0, stream>>>(lptr(br, l, rev, 0), WIH, 1024, Kin, Kp);
        cvt_pad_bf16<<<cdiv(1024L * 256, 256), 256, 0, stream>>>(lptr(br, l, rev, 1), WHH, 1024, 256, 256);
        add_vec<<<4, 256, 0, stream>>>(lptr(br, l, rev, 2), lptr(br, l, rev, 3), BIASF, 1024);
        dim3 g1(NN / 64, 1024 / 128);
        gemm_bf16<<<g1, 256, 0, stream>>>(ABUF, WIH, XG, BIASF, NN, 1024, Kp, 0);
        lstm_scan<<<BGN / 16, 512, 0, stream>>>(XG, WHH, LOUT, rev, rev ? 256 : 0);
      }
    }

    // lin: l2g = relu(wf @ lin_w^T + lin_b)
    cvt_pad_bf16<<<cdiv((long)NN * 512, 256), 256, 0, stream>>>(LOUT, ABUF, NN, 512, 512);
    cvt_pad_bf16<<<cdiv(256L * 512, 256), 256, 0, stream>>>(lin_w, WT, 256, 512, 512);
    {
      dim3 g(NN / 64, 2);
      gemm_bf16<<<g, 256, 0, stream>>>(ABUF, WT, L2G, lin_b, NN, 256, 512, 1);
    }

    // GCN 1
    cvt_transpose_bf16<<<cdiv(512L * 256, 256), 256, 0, stream>>>(gw1[br], WT, 512, 256);
    scale_cvt_bf16<<<cdiv((long)NN * 512, 256), 256, 0, stream>>>(LOUT, DEGO, ABUF, NN, 512);
    {
      dim3 g(NN / 64, 2);
      gemm_bf16<<<g, 256, 0, stream>>>(ABUF, WT, T6, nullptr, NN, 256, 512, 0);
    }
    zero_f32<<<cdiv((long)NN * 256, 256), 256, 0, stream>>>(T7, (long)NN * 256);
    scatter_add<<<cdiv((long)E * 64, 256), 256, 0, stream>>>(src, dst, T6, T7, E);
    gcn_combine<<<cdiv((long)NN * 256, 256), 256, 0, stream>>>(T7, DEGI, gb1[br], L2G, T6, (long)NN * 256);

    // GCN 2
    cvt_transpose_bf16<<<cdiv(256L * 256, 256), 256, 0, stream>>>(gw2[br], WT, 256, 256);
    scale_cvt_bf16<<<cdiv((long)NN * 256, 256), 256, 0, stream>>>(T6, DEGO, ABUF, NN, 256);
    {
      dim3 g(NN / 64, 2);
      gemm_bf16<<<g, 256, 0, stream>>>(ABUF, WT, T7, nullptr, NN, 256, 256, 0);
    }
    zero_f32<<<cdiv((long)NN * 256, 256), 256, 0, stream>>>(T6, (long)NN * 256);
    scatter_add<<<cdiv((long)E * 64, 256), 256, 0, stream>>>(src, dst, T7, T6, E);
    gcn_combine<<<cdiv((long)NN * 256, 256), 256, 0, stream>>>(T6, DEGI, gb2[br], L2G, T7, (long)NN * 256);

    graph_mean<<<cdiv(BGN * 256L, 256), 256, 0, stream>>>(T7, REP[br]);
  }

  pred_kernel<<<cdiv(BGN, 256), 256, 0, stream>>>(REP[0], REP[1], pred_w, pred_b, (float*)d_out);
}
